// Block_66073776882206
// MI455X (gfx1250) — matrix operand order repaired and checked by reading, not compiled, not run on hardware
//
#include <hip/hip_runtime.h>
#include <cstddef>
#include <cstdint>

// Problem constants (from the reference)
#define BB       500   // batch
#define BPAD     512   // batch padded to 32 tiles of 16 rows
#define TT       24    // BACKLEN
#define NVC      8     // input channels
#define FORECAST 12

// ---------------- WMMA types (CDNA5 / gfx1250, wave32) ----------------
typedef __attribute__((ext_vector_type(16))) __bf16 v16bf;
typedef __attribute__((ext_vector_type(8)))  float  v8f;

// ---------------- device helpers ----------------
__device__ __forceinline__ unsigned short f2bf(float f) {
  union { float f; unsigned u; } v; v.f = f;
  unsigned r = (v.u + 0x7FFFu + ((v.u >> 16) & 1u)) >> 16;  // RNE
  return (unsigned short)r;
}
__device__ __forceinline__ float bf2f(unsigned short h) {
  union { unsigned u; float f; } v; v.u = ((unsigned)h) << 16;
  return v.f;
}
__device__ __forceinline__ float sigf(float x) { return 1.f / (1.f + __expf(-x)); }

union TileU { struct { uint4 lo, hi; } q; v16bf v; };

// A-operand tile: 16x32 bf16 from row-major activations (ld multiple of 8).
// ISA 7.12.2 16-bit A layout: lanes 0-15 (half=0) hold K {0..7,16..23},
// lanes 16-31 (half=1) hold K {8..15,24..31}; row M = lane%16.
__device__ __forceinline__ v16bf load_a_tile(const unsigned short* p0, int ld) {
  const int lane = threadIdx.x & 31;
  const int half = lane >> 4, m = lane & 15;
  const unsigned short* p = p0 + (size_t)m * ld + 8 * half;
  TileU u;
  u.q.lo = *(const uint4*)p;         // K = 8*half .. 8*half+7
  u.q.hi = *(const uint4*)(p + 16);  // K = 16+8*half .. 16+8*half+7
  return u.v;
}
// B-operand tile: 32x16 bf16 taken from W stored row-major as (N rows x K cols):
// lane group g = lane/16 holds K {8g..8g+7, 16+8g..16+8g+7} for column n = lane%16.
__device__ __forceinline__ v16bf load_b_tile(const unsigned short* p0, int ld) {
  const int lane = threadIdx.x & 31;
  const int g = lane >> 4, n = lane & 15;
  const unsigned short* p = p0 + (size_t)n * ld + 8 * g;
  TileU u;
  u.q.lo = *(const uint4*)p;         // K = 8g .. 8g+7
  u.q.hi = *(const uint4*)(p + 16);  // K = 16+8g .. 16+8g+7
  return u.v;
}

// ---------------- persistent batch-tile LSTM sequence kernel ----------------
// One workgroup = 16 batch rows, 8 wave32 waves. Loops T steps internally.
// Gates (16 x 4H, f32) in LDS; h (16 x Hpad, bf16, A-layout) in LDS; c in regs.
__global__ void __launch_bounds__(256)
lstm_seq_kernel(const unsigned short* __restrict__ X, int XLD, int XKT,
                const unsigned short* __restrict__ Wi,  // (4H x XKT*32) bf16 row-major
                const unsigned short* __restrict__ Wh,  // (4H x Hpad)   bf16 row-major
                int HKT, const float* __restrict__ bias, int H, int NT,
                unsigned short* __restrict__ Y, int YLD, int ycol0,
                const int* __restrict__ tmap, int Tsteps,
                const unsigned short* __restrict__ Hin, const float* __restrict__ Cin,
                unsigned short* __restrict__ Hout, float* __restrict__ Cout) {
  const int Hpad = HKT * 32;
  const int G4 = 4 * H;  // == NT*16
  extern __shared__ char smem[];
  float* gates = (float*)smem;                                        // 16*G4 f32
  unsigned short* hbuf = (unsigned short*)(smem + (size_t)16 * G4 * 4); // 16*Hpad bf16
  const int tid = threadIdx.x;
  const int rb = blockIdx.x * 16;

  for (int i = tid; i < 16 * Hpad; i += 256) {
    int r = i / Hpad, c = i % Hpad;
    unsigned short v = 0;
    if (Hin && c < H) v = Hin[(size_t)(rb + r) * Hpad + c];
    hbuf[i] = v;
  }
  float creg[13];
#pragma unroll
  for (int j = 0; j < 13; ++j) creg[j] = 0.f;
  if (Cin) {
    for (int idx = tid, j = 0; idx < 16 * H; idx += 256, ++j) {
      int r = idx / H, c = idx % H;
      creg[j] = Cin[(size_t)(rb + r) * Hpad + c];
    }
  }
  __syncthreads();

  const int wave = tid >> 5;
  const int lane = tid & 31;
  const int bg = lane >> 4, bn = lane & 15;
  const int WiLD = XKT * 32;

  for (int s = 0; s < Tsteps; ++s) {
    const int tio = tmap ? tmap[s] : s;
    const unsigned short* Xs = X + ((size_t)tio * BPAD + rb) * XLD;
    // gate GEMM: 16 x 4H = sum_k A(16x32) * B(32x16), bf16 WMMA, f32 accum
    for (int nt = wave; nt < NT; nt += 8) {
      v8f acc = {0.f, 0.f, 0.f, 0.f, 0.f, 0.f, 0.f, 0.f};
      for (int kt = 0; kt < XKT; ++kt) {
        v16bf a = load_a_tile(Xs + kt * 32, XLD);
        v16bf b = load_b_tile(Wi + (size_t)(nt * 16) * WiLD + kt * 32, WiLD);
        acc = __builtin_amdgcn_wmma_f32_16x16x32_bf16(false, a, false, b,
                                                      (short)0, acc, false, false);
      }
      for (int kt = 0; kt < HKT; ++kt) {
        v16bf a = load_a_tile(hbuf + kt * 32, Hpad);
        v16bf b = load_b_tile(Wh + (size_t)(nt * 16) * Hpad + kt * 32, Hpad);
        acc = __builtin_amdgcn_wmma_f32_16x16x32_bf16(false, a, false, b,
                                                      (short)0, acc, false, false);
      }
      // D layout (ISA 7.12.2): VGPR v -> M = v + 8*(lane/16), N = lane%16
#pragma unroll
      for (int v = 0; v < 8; ++v)
        gates[(size_t)(v + 8 * bg) * G4 + nt * 16 + bn] = acc[v];
    }
    __syncthreads();
    // elementwise LSTM cell
    for (int idx = tid, j = 0; idx < 16 * H; idx += 256, ++j) {
      int r = idx / H, c = idx % H;
      const float* gr = gates + (size_t)r * G4;
      float gi = gr[c] + bias[c];
      float gf = gr[H + c] + bias[H + c];
      float gg = gr[2 * H + c] + bias[2 * H + c];
      float go = gr[3 * H + c] + bias[3 * H + c];
      float cv = sigf(gf) * creg[j] + sigf(gi) * tanhf(gg);
      creg[j] = cv;
      unsigned short hb = f2bf(sigf(go) * tanhf(cv));
      hbuf[r * Hpad + c] = hb;
      Y[((size_t)tio * BPAD + rb + r) * YLD + ycol0 + c] = hb;
    }
    __syncthreads();
  }
  if (Hout || Cout) {
    for (int idx = tid, j = 0; idx < 16 * H; idx += 256, ++j) {
      int r = idx / H, c = idx % H;
      if (Cout) Cout[(size_t)(rb + r) * Hpad + c] = creg[j];
      if (Hout) Hout[(size_t)(rb + r) * Hpad + c] = hbuf[r * Hpad + c];
    }
  }
}

// ---------------- small support kernels ----------------
__global__ void tmaps_kernel(int* __restrict__ maps) {
  int idx = blockIdx.x * blockDim.x + threadIdx.x;
  if (idx >= TT * TT) return;
  int f = idx / TT, s = idx % TT;
  maps[idx] = (s < f) ? (f - 1 - s) : (TT - 1 - (s - f));  // self-inverse perm
}

__global__ void pack_weight_kernel(const float* __restrict__ src, unsigned short* __restrict__ dst,
                                   int rows, int cols, int Kpad, int seg, int segoff) {
  long idx = (long)blockIdx.x * blockDim.x + threadIdx.x;
  if (idx >= (long)rows * Kpad) return;
  int r = (int)(idx / Kpad), k = (int)(idx % Kpad);
  int c = -1;
  if (k < seg) c = k;
  else if (k >= segoff) c = k - segoff + seg;
  float v = 0.f;
  if (c >= 0 && c < cols) v = src[(size_t)r * cols + c];
  dst[idx] = f2bf(v);
}

__global__ void pack_x_kernel(const float* __restrict__ xt, unsigned short* __restrict__ dst,
                              int z1, int z2) {
  long idx = (long)blockIdx.x * blockDim.x + threadIdx.x;
  if (idx >= (long)TT * BPAD * 32) return;
  int k = (int)(idx & 31);
  int r = (int)((idx >> 5) % BPAD);
  int t = (int)(idx / ((long)BPAD * 32));
  float v = 0.f;
  if (r < BB && k < NVC) v = xt[((size_t)r * TT + t) * NVC + k];
  if ((k == 1 && z1) || (k == 2 && z2)) v = 0.f;
  dst[idx] = f2bf(v);
}

__global__ void masks_kernel(const float* __restrict__ xt, const float* __restrict__ xorig,
                             float* __restrict__ inds, float* __restrict__ Cinds,
                             float* __restrict__ both, float* __restrict__ m1,
                             float* __restrict__ m2) {
  int r = blockIdx.x * blockDim.x + threadIdx.x;
  if (r >= BPAD) return;
  float s1 = 0.f, s2 = 0.f;
  for (int t = 0; t < TT; ++t) {
    float a = (r < BB) ? xt[((size_t)r * TT + t) * NVC + 1] : 0.f;
    float b = (r < BB) ? xt[((size_t)r * TT + t) * NVC + 2] : 0.f;
    s1 += a; s2 += b;
    m1[(size_t)r * TT + t] = (r < BB && xorig[((size_t)r * TT + t) * NVC + 1] > 0.f) ? 1.f : 0.f;
    m2[(size_t)r * TT + t] = (r < BB && xorig[((size_t)r * TT + t) * NVC + 2] > 0.f) ? 1.f : 0.f;
  }
  inds[r]  = (r < BB && s1 > 0.f) ? 1.f : 0.f;
  Cinds[r] = (r < BB && s2 > 0.f) ? 1.f : 0.f;
  both[r]  = (r < BB && (s1 + s2) > 0.f) ? 1.f : 0.f;
}

__global__ void acc_masked_kernel(const unsigned short* __restrict__ Ylast, int YLD, int Hpad,
                                  int H, const float* __restrict__ mask, int f,
                                  float* __restrict__ acc) {
  int idx = blockIdx.x * blockDim.x + threadIdx.x;
  if (idx >= BPAD * 2 * H) return;
  int r = idx / (2 * H), c = idx % (2 * H);
  int col = (c < H) ? c : (c - H + Hpad);
  acc[idx] += mask[(size_t)r * TT + f] * bf2f(Ylast[(size_t)r * YLD + col]);
}

__global__ void extract_kernel(const unsigned short* __restrict__ Ylast, int YLD, int Hpad,
                               int H, float* __restrict__ dst) {
  int idx = blockIdx.x * blockDim.x + threadIdx.x;
  if (idx >= BPAD * 2 * H) return;
  int r = idx / (2 * H), c = idx % (2 * H);
  int col = (c < H) ? c : (c - H + Hpad);
  dst[idx] = bf2f(Ylast[(size_t)r * YLD + col]);
}

__global__ void concat2_bf16_kernel(const float* __restrict__ A, int wA,
                                    const float* __restrict__ Bp, int wB,
                                    unsigned short* __restrict__ dst, int LD) {
  long idx = (long)blockIdx.x * blockDim.x + threadIdx.x;
  if (idx >= (long)BPAD * LD) return;
  int r = (int)(idx / LD), k = (int)(idx % LD);
  float v = 0.f;
  if (k < wA) v = A[(size_t)r * wA + k];
  else if (k < wA + wB) v = Bp[(size_t)r * wB + (k - wA)];
  dst[idx] = f2bf(v);
}

// decoder per-step gather: yx correction, carried s/sc, two linear heads
__global__ void dec_post_kernel(const unsigned short* __restrict__ Yd,   // (BPAD,256)
                                const unsigned short* __restrict__ Ys,   // (BPAD,448)
                                const unsigned short* __restrict__ Ysc,  // (BPAD,448)
                                const float* __restrict__ inds, const float* __restrict__ Cinds,
                                const float* __restrict__ both,
                                const float* __restrict__ lin_w, const float* __restrict__ lin_b,
                                const float* __restrict__ linS_w, const float* __restrict__ linS_b,
                                float* __restrict__ y, float* __restrict__ s,
                                float* __restrict__ sc, float* __restrict__ out, int step) {
  int r = blockIdx.x * blockDim.x + threadIdx.x;
  if (r >= BPAD) return;
  const unsigned short* yd = Yd + (size_t)r * 256;
  const unsigned short* ys = Ys + (size_t)r * 448;
  const unsigned short* ysc = Ysc + (size_t)r * 448;
  const float mi = inds[r], mc = Cinds[r];
  float dot = 0.f, sdot = 0.f;
  for (int c = 0; c < 200; ++c) {
    float ydv  = bf2f(yd[(c < 100) ? c : (c + 28)]);  // fwd cols 0..99, bwd at 128..227
    float ysl  = bf2f(ys[c]);                          // yS[:, :H2]
    float ysu  = bf2f(ys[224 + c]);                    // yS[:, H2:]
    float yscl = bf2f(ysc[c]);
    float yscu = bf2f(ysc[224 + c]);
    float yx = ydv;
    if (mi != 0.f) yx -= fmaxf(ysl, 0.f);
    if (mc != 0.f) yx += fmaxf(yscl, 0.f);
    dot += yx * lin_w[c];
    float sT  = (mi != 0.f) ? ysu : 0.f;
    float scT = (mc != 0.f) ? yscu : 0.f;
    sdot += (sT + scT) * linS_w[c];
    y[(size_t)r * 200 + c] = yx;
    s[(size_t)r * 200 + c] = ysu;
    sc[(size_t)r * 200 + c] = yscu;
  }
  if (r < BB) {
    out[(size_t)r * FORECAST + step] = dot + lin_b[0];
    out[(size_t)BB * FORECAST + (size_t)r * FORECAST + step] =
        (both[r] != 0.f) ? (sdot + linS_b[0]) : 0.f;
  }
}

// ---------------- host-side orchestration ----------------
static inline int nblk(long n) { return (int)((n + 255) / 256); }
static inline int rup32(int n) { return (n + 31) & ~31; }

struct Arena {
  char* base; size_t off;
  void* take(size_t bytes) {
    off = (off + 255) & ~(size_t)255;
    void* p = base + off; off += bytes; return p;
  }
};

struct PackedLSTM {
  unsigned short *Wi0[2], *Wh0[2], *Wi1[2], *Wh1[2];
  const float *b0, *b1;
  int K0, H, Hpad, HKT;
};

static PackedLSTM pack_lstm(hipStream_t st, Arena& A, void* const* din, int base, int I0, int H) {
  PackedLSTM P;
  P.H = H; P.Hpad = rup32(H); P.HKT = P.Hpad / 32; P.K0 = rup32(I0);
  const int r4 = 4 * H;
  const float* Wih0 = (const float*)din[base + 0];
  const float* Whh0 = (const float*)din[base + 1];
  P.b0 = (const float*)din[base + 2];
  const float* Wih1 = (const float*)din[base + 3];
  const float* Whh1 = (const float*)din[base + 4];
  P.b1 = (const float*)din[base + 5];
  for (int d = 0; d < 2; ++d) {
    long n;
    n = (long)r4 * P.K0;
    P.Wi0[d] = (unsigned short*)A.take((size_t)n * 2);
    pack_weight_kernel<<<nblk(n), 256, 0, st>>>(Wih0 + (size_t)d * r4 * I0, P.Wi0[d],
                                                r4, I0, P.K0, I0, P.K0);
    n = (long)r4 * P.Hpad;
    P.Wh0[d] = (unsigned short*)A.take((size_t)n * 2);
    pack_weight_kernel<<<nblk(n), 256, 0, st>>>(Whh0 + (size_t)d * r4 * H, P.Wh0[d],
                                                r4, H, P.Hpad, H, P.Hpad);
    n = (long)r4 * 2 * P.Hpad;  // split layout: cols [0,H) -> 0.., [H,2H) -> Hpad..
    P.Wi1[d] = (unsigned short*)A.take((size_t)n * 2);
    pack_weight_kernel<<<nblk(n), 256, 0, st>>>(Wih1 + (size_t)d * r4 * (2 * H), P.Wi1[d],
                                                r4, 2 * H, 2 * P.Hpad, H, P.Hpad);
    n = (long)r4 * P.Hpad;
    P.Wh1[d] = (unsigned short*)A.take((size_t)n * 2);
    pack_weight_kernel<<<nblk(n), 256, 0, st>>>(Whh1 + (size_t)d * r4 * H, P.Wh1[d],
                                                r4, H, P.Hpad, H, P.Hpad);
  }
  return P;
}

static void launch_seq(hipStream_t st, const unsigned short* Xp, int XLD, int XKT,
                       const PackedLSTM& P, int layer, int dir,
                       unsigned short* Y, int YLD, int ycol0,
                       const int* tmap, int Tsteps,
                       const unsigned short* Hin, const float* Cin,
                       unsigned short* Hout, float* Cout) {
  const unsigned short* Wi = (layer == 0) ? P.Wi0[dir] : P.Wi1[dir];
  const unsigned short* Wh = (layer == 0) ? P.Wh0[dir] : P.Wh1[dir];
  const float* bias = ((layer == 0) ? P.b0 : P.b1) + (size_t)dir * 4 * P.H;
  int NT = (4 * P.H) / 16;
  size_t smem = (size_t)16 * 4 * P.H * sizeof(float) + (size_t)16 * P.Hpad * 2;
  lstm_seq_kernel<<<BPAD / 16, 256, smem, st>>>(Xp, XLD, XKT, Wi, Wh, P.HKT, bias, P.H, NT,
                                                Y, YLD, ycol0, tmap, Tsteps,
                                                Hin, Cin, Hout, Cout);
}

static void shift_acc(hipStream_t st, const unsigned short* Xv, const PackedLSTM& P,
                      const float* mask, float* acc,
                      unsigned short* Y1, unsigned short* Y2, const int* tmaps) {
  hipMemsetAsync(acc, 0, (size_t)BPAD * 2 * P.H * sizeof(float), st);
  const int LD = 2 * P.Hpad;
  // yf1 is f-independent: compute fwd layer-0 once into Y1 cols [0,H)
  launch_seq(st, Xv, 32, 1, P, 0, 0, Y1, LD, 0, nullptr, TT, nullptr, nullptr, nullptr, nullptr);
  for (int f = 0; f < TT; ++f) {
    const int* map = tmaps + (size_t)f * TT;
    launch_seq(st, Xv, 32, 1, P, 0, 1, Y1, LD, P.Hpad, map, TT, nullptr, nullptr, nullptr, nullptr);
    launch_seq(st, Y1, LD, 2 * P.HKT, P, 1, 0, Y2, LD, 0, nullptr, TT, nullptr, nullptr, nullptr, nullptr);
    launch_seq(st, Y1, LD, 2 * P.HKT, P, 1, 1, Y2, LD, P.Hpad, map, TT, nullptr, nullptr, nullptr, nullptr);
    acc_masked_kernel<<<nblk((long)BPAD * 2 * P.H), 256, 0, st>>>(
        Y2 + (size_t)(TT - 1) * BPAD * LD, LD, P.Hpad, P.H, mask, f, acc);
  }
}

extern "C" void kernel_launch(void* const* d_in, const int* in_sizes, int n_in,
                              void* d_out, int out_size, void* d_ws, size_t ws_size,
                              hipStream_t stream) {
  (void)in_sizes; (void)n_in; (void)out_size; (void)ws_size;
  const float* xt     = (const float*)d_in[0];
  const float* xorig  = (const float*)d_in[1];
  const float* lin_w  = (const float*)d_in[38];
  const float* lin_b  = (const float*)d_in[39];
  const float* linS_w = (const float*)d_in[40];
  const float* linS_b = (const float*)d_in[41];
  float* out = (float*)d_out;

  Arena A{(char*)d_ws, 0};
  int*   tmaps = (int*)A.take((size_t)TT * TT * sizeof(int));
  float* inds  = (float*)A.take(BPAD * sizeof(float));
  float* Cinds = (float*)A.take(BPAD * sizeof(float));
  float* both  = (float*)A.take(BPAD * sizeof(float));
  float* m1    = (float*)A.take((size_t)BPAD * TT * sizeof(float));
  float* m2    = (float*)A.take((size_t)BPAD * TT * sizeof(float));
  unsigned short* Xsiv  = (unsigned short*)A.take((size_t)TT * BPAD * 32 * 2);
  unsigned short* XsivC = (unsigned short*)A.take((size_t)TT * BPAD * 32 * 2);
  unsigned short* Xin   = (unsigned short*)A.take((size_t)TT * BPAD * 32 * 2);
  unsigned short* Y1 = (unsigned short*)A.take((size_t)TT * BPAD * 256 * 2);
  unsigned short* Y2 = (unsigned short*)A.take((size_t)TT * BPAD * 256 * 2);
  float* accS  = (float*)A.take((size_t)BPAD * 200 * sizeof(float));
  float* accSC = (float*)A.take((size_t)BPAD * 200 * sizeof(float));
  float* ybuf  = (float*)A.take((size_t)BPAD * 200 * sizeof(float));
  unsigned short* xd  = (unsigned short*)A.take((size_t)BPAD * 224 * 2);
  unsigned short* xs  = (unsigned short*)A.take((size_t)BPAD * 416 * 2);
  unsigned short* xsc = (unsigned short*)A.take((size_t)BPAD * 416 * 2);
  unsigned short* Y01d = (unsigned short*)A.take((size_t)BPAD * 256 * 2);
  unsigned short* Y2d  = (unsigned short*)A.take((size_t)BPAD * 256 * 2);
  unsigned short* Y01s = (unsigned short*)A.take((size_t)BPAD * 448 * 2);
  unsigned short* Y2s  = (unsigned short*)A.take((size_t)BPAD * 448 * 2);
  unsigned short* Y01c = (unsigned short*)A.take((size_t)BPAD * 448 * 2);
  unsigned short* Y2c  = (unsigned short*)A.take((size_t)BPAD * 448 * 2);
  unsigned short *hd[4], *hs[4], *hc[4];
  float *cd[4], *cs[4], *cc[4];
  for (int i = 0; i < 4; ++i) {
    hd[i] = (unsigned short*)A.take((size_t)BPAD * 128 * 2);
    cd[i] = (float*)A.take((size_t)BPAD * 128 * sizeof(float));
    hs[i] = (unsigned short*)A.take((size_t)BPAD * 224 * 2);
    cs[i] = (float*)A.take((size_t)BPAD * 224 * sizeof(float));
    hc[i] = (unsigned short*)A.take((size_t)BPAD * 224 * 2);
    cc[i] = (float*)A.take((size_t)BPAD * 224 * sizeof(float));
  }

  // ---- setup: time maps, masks, bf16 input variants, bf16 packed weights ----
  tmaps_kernel<<<nblk(TT * TT), 256, 0, stream>>>(tmaps);
  masks_kernel<<<nblk(BPAD), 256, 0, stream>>>(xt, xorig, inds, Cinds, both, m1, m2);
  pack_x_kernel<<<nblk((long)TT * BPAD * 32), 256, 0, stream>>>(xt, Xsiv, 0, 1);  // zero ch2
  pack_x_kernel<<<nblk((long)TT * BPAD * 32), 256, 0, stream>>>(xt, XsivC, 1, 0); // zero ch1
  pack_x_kernel<<<nblk((long)TT * BPAD * 32), 256, 0, stream>>>(xt, Xin, 1, 1);   // zero both

  PackedLSTM Plstm   = pack_lstm(stream, A, d_in,  2, NVC, 100);
  PackedLSTM Pdec    = pack_lstm(stream, A, d_in,  8, 200, 100);
  PackedLSTM PlstmS  = pack_lstm(stream, A, d_in, 14, NVC, 100);
  PackedLSTM PdecS   = pack_lstm(stream, A, d_in, 20, 400, 200);
  PackedLSTM PlstmSC = pack_lstm(stream, A, d_in, 26, NVC, 100);
  PackedLSTM PdecSC  = pack_lstm(stream, A, d_in, 32, 400, 200);

  // ---- shift_accumulate (otemp / otempC) ----
  shift_acc(stream, Xsiv,  PlstmS,  m1, accS,  Y1, Y2, tmaps);
  shift_acc(stream, XsivC, PlstmSC, m2, accSC, Y1, Y2, tmaps);

  // ---- encoder: bilstm_seq(xin)[T-1] -> ybuf (compact f32, BPAD x 200) ----
  const int* rev = tmaps;  // f=0 map == plain reverse
  launch_seq(stream, Xin, 32, 1, Plstm, 0, 0, Y1, 256, 0, nullptr, TT, nullptr, nullptr, nullptr, nullptr);
  launch_seq(stream, Xin, 32, 1, Plstm, 0, 1, Y1, 256, 128, rev, TT, nullptr, nullptr, nullptr, nullptr);
  launch_seq(stream, Y1, 256, 8, Plstm, 1, 0, Y2, 256, 0, nullptr, TT, nullptr, nullptr, nullptr, nullptr);
  launch_seq(stream, Y1, 256, 8, Plstm, 1, 1, Y2, 256, 128, rev, TT, nullptr, nullptr, nullptr, nullptr);
  extract_kernel<<<nblk((long)BPAD * 200), 256, 0, stream>>>(
      Y2 + (size_t)(TT - 1) * BPAD * 256, 256, 128, 100, ybuf);

  // ---- decoder: 12 steps, state carried through ws (zero-init via null at step 0) ----
  for (int step = 0; step < FORECAST; ++step) {
    concat2_bf16_kernel<<<nblk((long)BPAD * 224), 256, 0, stream>>>(ybuf, 200, nullptr, 0, xd, 224);
    concat2_bf16_kernel<<<nblk((long)BPAD * 416), 256, 0, stream>>>(ybuf, 200, accS, 200, xs, 416);
    concat2_bf16_kernel<<<nblk((long)BPAD * 416), 256, 0, stream>>>(ybuf, 200, accSC, 200, xsc, 416);
    const bool f0 = (step == 0);
#define ST(h, c) (f0 ? nullptr : (h)), (f0 ? nullptr : (c)), (h), (c)
    // p_dec bilstm_step
    launch_seq(stream, xd, 224, 7, Pdec, 0, 0, Y01d, 256, 0,   nullptr, 1, ST(hd[0], cd[0]));
    launch_seq(stream, xd, 224, 7, Pdec, 0, 1, Y01d, 256, 128, nullptr, 1, ST(hd[1], cd[1]));
    launch_seq(stream, Y01d, 256, 8, Pdec, 1, 0, Y2d, 256, 0,   nullptr, 1, ST(hd[2], cd[2]));
    launch_seq(stream, Y01d, 256, 8, Pdec, 1, 1, Y2d, 256, 128, nullptr, 1, ST(hd[3], cd[3]));
    // p_decS bilstm_step (H=200)
    launch_seq(stream, xs, 416, 13, PdecS, 0, 0, Y01s, 448, 0,   nullptr, 1, ST(hs[0], cs[0]));
    launch_seq(stream, xs, 416, 13, PdecS, 0, 1, Y01s, 448, 224, nullptr, 1, ST(hs[1], cs[1]));
    launch_seq(stream, Y01s, 448, 14, PdecS, 1, 0, Y2s, 448, 0,   nullptr, 1, ST(hs[2], cs[2]));
    launch_seq(stream, Y01s, 448, 14, PdecS, 1, 1, Y2s, 448, 224, nullptr, 1, ST(hs[3], cs[3]));
    // p_decSC bilstm_step (H=200)
    launch_seq(stream, xsc, 416, 13, PdecSC, 0, 0, Y01c, 448, 0,   nullptr, 1, ST(hc[0], cc[0]));
    launch_seq(stream, xsc, 416, 13, PdecSC, 0, 1, Y01c, 448, 224, nullptr, 1, ST(hc[1], cc[1]));
    launch_seq(stream, Y01c, 448, 14, PdecSC, 1, 0, Y2c, 448, 0,   nullptr, 1, ST(hc[2], cc[2]));
    launch_seq(stream, Y01c, 448, 14, PdecSC, 1, 1, Y2c, 448, 224, nullptr, 1, ST(hc[3], cc[3]));
#undef ST
    dec_post_kernel<<<nblk(BPAD), 256, 0, stream>>>(Y2d, Y2s, Y2c, inds, Cinds, both,
                                                    lin_w, lin_b, linS_w, linS_b,
                                                    ybuf, accS, accSC, out, step);
  }
}
